// scaledDotProductAttention_79577154061017
// MI455X (gfx1250) — matrix operand order repaired and checked by reading, not compiled, not run on hardware
//
#include <hip/hip_runtime.h>

typedef __attribute__((ext_vector_type(16))) _Float16 v16h;
typedef __attribute__((ext_vector_type(8)))  _Float16 v8h;
typedef __attribute__((ext_vector_type(8)))  float    v8f;
typedef __attribute__((ext_vector_type(4)))  float    v4f;

namespace {
constexpr int kS  = 2048;   // sequence length
constexpr int kD  = 128;    // head dim
constexpr int kBH = 32;     // batch * heads
constexpr int kPS = 2056;   // LDS row stride (halves) for score/P strip: 2048 + 8 pad, mult of 8
constexpr int kQS = 136;    // LDS row stride (halves) for Q tile: 128 + 8 pad
constexpr size_t kSmemBytes =
    (size_t)16 * kPS * 2 +   // score/P strip (f16, reused in place)
    (size_t)16 * kQS * 2 +   // Q tile (f16)
    256 * 4 +                // partial-reduction scratch [16][16]
    16 * 4 +                 // row max
    16 * 4;                  // 1/rowsum
}

// Build the 16x32 f16 A-fragment (per documented CDNA5 VGPR layout) from a
// row-major f16 tile in LDS. Lanes 0-15: M=lane, halves = K[kOff..kOff+7],
// K[kOff+16..kOff+23]; lanes 16-31: M=lane-16, +8 on each K range.
// Two 16-byte ds_load_b128 per lane.
__device__ __forceinline__ v16h load_a_frag(const _Float16* base, int strideH,
                                            int kOff, int lane) {
  const int m = lane & 15, hid = lane >> 4;
  const _Float16* p = base + m * strideH + kOff + 8 * hid;
  v8h lo = *(const v8h*)p;          // K = kOff + 8*hid + [0..7]
  v8h hi = *(const v8h*)(p + 16);   // K = kOff + 16 + 8*hid + [0..7]
  v16h a;
#pragma unroll
  for (int i = 0; i < 8; ++i) { a[i] = lo[i]; a[8 + i] = hi[i]; }
  return a;
}

__global__ __launch_bounds__(256)
void attn_fused(const float* __restrict__ Q, const float* __restrict__ K,
                const float* __restrict__ V, float* __restrict__ O,
                float* __restrict__ W) {
  extern __shared__ char smem[];
  _Float16* Sh = (_Float16*)smem;            // 16 x kPS : scores, then P (in place)
  _Float16* Qt = Sh + 16 * kPS;              // 16 x kQS : Q tile f16
  float* red    = (float*)(Qt + 16 * kQS);   // 16 x 16 partials
  float* rowmax = red + 256;                 // 16
  float* rinv   = rowmax + 16;               // 16

  const int tid  = threadIdx.x;
  const int lane = tid & 31;
  const int wave = tid >> 5;
  const int n    = lane & 15;   // WMMA B/D column within tile
  const int hid  = lane >> 4;   // half-wave id
  const int bh    = blockIdx.x;       // 0..31
  const int qb    = blockIdx.y;       // 0..127 query block
  const int qbase = qb << 4;

  const float* Qp = Q + ((size_t)bh * kS + qbase) * kD;
  const float* Kp = K + (size_t)bh * kS * kD;
  const float* Vp = V + (size_t)bh * kS * kD;
  float* Op = O + ((size_t)bh * kS + qbase) * kD;
  float* Wp = W + ((size_t)bh * kS + qbase) * kS;

  // Prefill score strip with the mask value; load Q tile as f16.
  for (int i = tid; i < 16 * kS; i += 256)
    Sh[(i >> 11) * kPS + (i & (kS - 1))] = (_Float16)(-10000.0f);
  for (int i = tid; i < 16 * kD; i += 256) {
    const int r = i >> 7, c = i & (kD - 1);
    Qt[r * kQS + c] = (_Float16)Qp[r * kD + c];
  }
  __syncthreads();

  const float scale = 0.088388347648318447f;  // 1/sqrt(128)

  // ---- S = scale * (Q K^T), causal: only K-blocks j <= qb are not fully masked
  for (int j = wave; j <= qb; j += 8) {
    const float* krow = Kp + (size_t)((j << 4) + n) * kD;  // this lane's K row
    v8f acc = {};
#pragma unroll
    for (int kk = 0; kk < kD; kk += 32) {
      v16h a = load_a_frag(Qt, kQS, kk, lane);
      // B-frag: lane holds column n, halves = K_dim [kk+8*hid .. +7],
      // [kk+16+8*hid .. +7], read as 4x float4 from global and convert inline to f16.
      const float* kp = krow + kk + 8 * hid;
      v16h b;
#pragma unroll
      for (int i = 0; i < 16; i += 4) {
        v4f t = *(const v4f*)(kp + i + (i & 8));
        b[i]     = (_Float16)t[0]; b[i + 1] = (_Float16)t[1];
        b[i + 2] = (_Float16)t[2]; b[i + 3] = (_Float16)t[3];
      }
      acc = __builtin_amdgcn_wmma_f32_16x16x32_f16(false, a, false, b,
                                                   (short)0, acc, false, false);
    }
    const int col = (j << 4) + n;
#pragma unroll
    for (int r = 0; r < 8; ++r) {
      const int row = r + (hid << 3);  // D layout: M = r + 8*hid, N = lane%16
      const float s = (col > qbase + row) ? -10000.0f : acc[r] * scale;
      Sh[row * kPS + col] = (_Float16)s;
    }
  }
  __syncthreads();

  // ---- softmax: segmented max / sum-exp (16 threads per row, 128 elems each)
  {
    const int r = tid >> 4, sg = tid & 15;
    const _Float16* p = Sh + r * kPS + (sg << 7);
    float m = -3.0e38f;
#pragma unroll 8
    for (int i = 0; i < 128; ++i) m = fmaxf(m, (float)p[i]);
    red[(r << 4) + sg] = m;
  }
  __syncthreads();
  if (tid < 16) {
    float m = red[tid << 4];
    for (int i = 1; i < 16; ++i) m = fmaxf(m, red[(tid << 4) + i]);
    rowmax[tid] = m;
  }
  __syncthreads();
  {
    const int r = tid >> 4, sg = tid & 15;
    const _Float16* p = Sh + r * kPS + (sg << 7);
    const float mx = rowmax[r];
    float s = 0.0f;
#pragma unroll 8
    for (int i = 0; i < 128; ++i) s += __expf((float)p[i] - mx);
    red[(r << 4) + sg] = s;
  }
  __syncthreads();
  if (tid < 16) {
    float s = 0.0f;
    for (int i = 0; i < 16; ++i) s += red[(tid << 4) + i];
    rinv[tid] = 1.0f / s;
  }
  __syncthreads();

  // ---- normalize in place (Sh becomes P in f16), stream weights to HBM.
  // Coalesced, write-once -> non-temporal so K/V stay hot in L2.
  for (int i = tid; i < 16 * kS; i += 256) {
    const int r = i >> 11, c = i & (kS - 1);
    const float p = __expf((float)Sh[r * kPS + c] - rowmax[r]) * rinv[r];
    __builtin_nontemporal_store(p, &Wp[(size_t)r * kS + c]);
    Sh[r * kPS + c] = (_Float16)p;
  }
  __syncthreads();

  // ---- O = P V ; wave owns output columns [16*wave, 16*wave+16).
  // Rows of V beyond the causal limit have exactly-zero weights: skip them.
  {
    int kLimit = (qbase + 16 + 31) & ~31;
    if (kLimit > kS) kLimit = kS;
    const int colBase = wave << 4;
    v8f acc = {};
    for (int kb = 0; kb < kLimit; kb += 32) {
      v16h a = load_a_frag(Sh, kPS, kb, lane);
      // B-frag: lane = column colBase+n; k = kb + 8*hid + e + (e&8). For fixed e the
      // 16 lanes read 16 consecutive floats of one V row -> coalesced.
      const float* vp = Vp + (size_t)(kb + 8 * hid) * kD + colBase + n;
      v16h b;
#pragma unroll
      for (int e = 0; e < 16; ++e) b[e] = (_Float16)vp[(size_t)(e + (e & 8)) * kD];
      acc = __builtin_amdgcn_wmma_f32_16x16x32_f16(false, a, false, b,
                                                   (short)0, acc, false, false);
    }
#pragma unroll
    for (int r = 0; r < 8; ++r) {
      __builtin_nontemporal_store(
          acc[r], &Op[(size_t)(r + (hid << 3)) * kD + colBase + n]);
    }
  }
}

extern "C" void kernel_launch(void* const* d_in, const int* in_sizes, int n_in,
                              void* d_out, int out_size, void* d_ws, size_t ws_size,
                              hipStream_t stream) {
  (void)in_sizes; (void)n_in; (void)d_ws; (void)ws_size; (void)out_size;
  const float* q = (const float*)d_in[0];
  const float* k = (const float*)d_in[1];
  const float* v = (const float*)d_in[2];
  // d_in[3] is the boolean causal mask; the kernel applies causality analytically.
  float* out = (float*)d_out;                       // [B,H,S,D]
  float* w   = out + (size_t)kBH * kS * kD;         // [B,H,S,S] follows in d_out

  hipFuncSetAttribute((const void*)attn_fused,
                      hipFuncAttributeMaxDynamicSharedMemorySize, (int)kSmemBytes);
  dim3 grid(kBH, kS / 16);
  attn_fused<<<grid, dim3(256), kSmemBytes, stream>>>(q, k, v, out, w);
}
